// Block_3908420239427
// MI455X (gfx1250) — matrix operand order repaired and checked by reading, not compiled, not run on hardware
//
#include <hip/hip_runtime.h>
#include <hip/hip_bf16.h>
#include <math.h>

// ---------------------------------------------------------------------------
// Types for CDNA5 WMMA (gfx1250, wave32)
// ---------------------------------------------------------------------------
typedef __attribute__((ext_vector_type(16))) __bf16          v16bf;
typedef __attribute__((ext_vector_type(8)))  float           v8f;
typedef __attribute__((ext_vector_type(4)))  unsigned short  u16x4;
typedef __attribute__((ext_vector_type(8)))  unsigned short  u16x8;
typedef __attribute__((ext_vector_type(16))) unsigned short  u16x16;

// fp32 -> bf16 RNE via hardware convert (clang lowers the cast to
// v_cvt_*bf16_f32 on gfx1250 instead of a 3-op integer sequence)
__device__ __forceinline__ unsigned short f2bf(float f) {
    return __builtin_bit_cast(unsigned short, (__bf16)f);
}

// Build a 16-element bf16 fragment from two 8-element (16-byte) LDS loads.
__device__ __forceinline__ v16bf frag16(const unsigned short* p0,
                                        const unsigned short* p1) {
    u16x8 lo = *(const u16x8*)p0;
    u16x8 hi = *(const u16x8*)p1;
    u16x16 c;
#pragma unroll
    for (int i = 0; i < 8; ++i) { c[i] = lo[i]; c[i + 8] = hi[i]; }
    return __builtin_bit_cast(v16bf, c);
}

__device__ __forceinline__ float gelu_f(float x) {
    float x3 = x * x * x;
    float t  = tanhf(0.7978845608028654f * (x + 0.044715f * x3));
    return 0.5f * x * (1.0f + t);
}

// ---------------------------------------------------------------------------
// GEMM:  C[N x M] = A[N x K] @ W[K x M] + bias   (optional GELU)
// Block: 256 threads (8 waves), tile 128(N) x 128(M), K-step 64.
// Wave grid 2x4 -> each wave computes 64x32 via 4x2 wmma_f32_16x16x32_bf16
// per 32-deep K chunk (16 WMMAs per K-step).
// Register double-buffering: K-tile kt+1 is loaded to VGPRs while kt computes;
// kt+2 is warmed into GL2 with global_prefetch_b8.
// W staging: each thread owns 8 consecutive k-rows of a 4-column quad so the
// transposed tile commits as four 16-byte u16x8 LDS stores per thread.
// All dims are multiples of 128/64 in this problem; no bounds checks.
// ---------------------------------------------------------------------------
template <bool GELU>
__global__ __launch_bounds__(256)
void gemm_bias(const float* __restrict__ A, const float* __restrict__ W,
               const float* __restrict__ bias, float* __restrict__ C,
               int K, int M) {
    constexpr int SA = 72;   // LDS row stride (elements); 144B rows, 16B aligned
    __shared__ unsigned short sA[128 * SA];   // [row][k]   k = 0..63
    __shared__ unsigned short sB[128 * SA];   // [col][k]   (W tile transposed)

    const int tid  = threadIdx.x;
    const int lane = tid & 31;
    const int wave = tid >> 5;
    const int wr   = wave >> 2;      // 0..1 : 64 rows
    const int wc   = wave & 3;       // 0..3 : 32 cols
    const int r16  = lane & 15;
    const int half = lane >> 4;

    const int rowBase = blockIdx.y * 128;
    const int colBase = blockIdx.x * 128;

    // per-thread global-load coordinates
    const int aRow0 = tid >> 4;          // + it*16, it<8  -> rows 0..127
    const int aKq   = (tid & 15) * 4;    // k quad within 64
    const int bK0   = (tid >> 5) * 8;    // 8 consecutive k-rows per thread
    const int bCol  = (tid & 31) * 4;    // col quad within 128

    float4 ra[8], rb[8];
    const float* rbf = (const float*)rb;

    auto loadTiles = [&](int kBase) {
#pragma unroll
        for (int it = 0; it < 8; ++it) {
            int row = aRow0 + it * 16;
            ra[it] = *(const float4*)(A + (size_t)(rowBase + row) * K + kBase + aKq);
        }
#pragma unroll
        for (int it = 0; it < 8; ++it) {
            int krow = bK0 + it;
            rb[it] = *(const float4*)(W + (size_t)(kBase + krow) * M + colBase + bCol);
        }
    };
    auto stageTiles = [&]() {
#pragma unroll
        for (int it = 0; it < 8; ++it) {
            int row = aRow0 + it * 16;
            u16x4 pk = {f2bf(ra[it].x), f2bf(ra[it].y), f2bf(ra[it].z), f2bf(ra[it].w)};
            *(u16x4*)(&sA[row * SA + aKq]) = pk;
        }
#pragma unroll
        for (int c = 0; c < 4; ++c) {
            u16x8 pk = {f2bf(rbf[0 * 4 + c]), f2bf(rbf[1 * 4 + c]),
                        f2bf(rbf[2 * 4 + c]), f2bf(rbf[3 * 4 + c]),
                        f2bf(rbf[4 * 4 + c]), f2bf(rbf[5 * 4 + c]),
                        f2bf(rbf[6 * 4 + c]), f2bf(rbf[7 * 4 + c])};
            *(u16x8*)(&sB[(bCol + c) * SA + bK0]) = pk;   // k = bK0..bK0+7
        }
    };

    v8f acc[4][2];
#pragma unroll
    for (int mt = 0; mt < 4; ++mt)
#pragma unroll
        for (int nt = 0; nt < 2; ++nt)
            acc[mt][nt] = (v8f){0.f, 0.f, 0.f, 0.f, 0.f, 0.f, 0.f, 0.f};

    loadTiles(0);
    stageTiles();

    const int ksteps = K >> 6;
    for (int kt = 0; kt < ksteps; ++kt) {
        __syncthreads();                       // staged LDS tile ready
        const bool more = (kt + 1) < ksteps;
        if (more) loadTiles((kt + 1) * 64);    // overlap global latency with WMMA
        if (kt + 2 < ksteps) {                 // warm GL2 two steps ahead
            __builtin_prefetch(A + (size_t)(rowBase + aRow0) * K + (kt + 2) * 64 + aKq, 0, 1);
            __builtin_prefetch(W + (size_t)((kt + 2) * 64 + bK0) * M + colBase + bCol, 0, 1);
        }

#pragma unroll
        for (int kc = 0; kc < 2; ++kc) {
            v16bf afrag[4];
#pragma unroll
            for (int mt = 0; mt < 4; ++mt) {
                int row = wr * 64 + mt * 16 + r16;
                const unsigned short* p = &sA[row * SA + kc * 32 + 8 * half];
                afrag[mt] = frag16(p, p + 16);
            }
            v16bf bfrag[2];
#pragma unroll
            for (int nt = 0; nt < 2; ++nt) {
                int col = wc * 32 + nt * 16 + r16;
                const unsigned short* p = &sB[col * SA + kc * 32 + 8 * half];
                bfrag[nt] = frag16(p, p + 16);
            }
#pragma unroll
            for (int mt = 0; mt < 4; ++mt)
#pragma unroll
                for (int nt = 0; nt < 2; ++nt)
                    acc[mt][nt] = __builtin_amdgcn_wmma_f32_16x16x32_bf16(
                        false, afrag[mt], false, bfrag[nt],
                        (short)0, acc[mt][nt], false, false);
        }
        __syncthreads();                       // all waves done reading LDS
        if (more) stageTiles();                // commit kt+1 to LDS
    }

    // --- epilogue: bias (+GELU), store ---
#pragma unroll
    for (int mt = 0; mt < 4; ++mt) {
#pragma unroll
        for (int nt = 0; nt < 2; ++nt) {
            int col = colBase + wc * 32 + nt * 16 + r16;
            float bcol = bias[col];
#pragma unroll
            for (int v = 0; v < 8; ++v) {
                int row = rowBase + wr * 64 + mt * 16 + v + 8 * half;
                float val = acc[mt][nt][v] + bcol;
                if (GELU) val = gelu_f(val);
                C[(size_t)row * M + col] = val;
            }
        }
    }
}

// ---------------------------------------------------------------------------
// Flash attention, causal.  qkv: [4096 tokens x 3072] (q|k|v, col = h*64+dh)
// out a0: [4096 x 1024].  Grid (qt=32, bh=32), block = 128 threads (4 waves).
// Each wave owns 16 query rows; workgroup handles a 64-query tile.
// ---------------------------------------------------------------------------
__global__ __launch_bounds__(128)
void attn_kernel(const float* __restrict__ qkv, float* __restrict__ out) {
    constexpr int S = 2048, D = 1024, QKV = 3072, ST = 72;  // 144B rows, 16B aligned
    const int tid  = threadIdx.x;
    const int lane = tid & 31;
    const int wave = tid >> 5;       // 0..3
    const int r16  = lane & 15;
    const int half = lane >> 4;

    const int qt = blockIdx.x;       // query tile (64 rows)
    const int bh = blockIdx.y;
    const int b  = bh >> 4;
    const int h  = bh & 15;

    __shared__ unsigned short sQ[64 * ST];    // [qrow][d]
    __shared__ unsigned short sK[64 * ST];    // [key][d]   (== B layout [N][K])
    __shared__ unsigned short sVT[64 * ST];   // [dh][key]  (V transposed)
    __shared__ unsigned short sP[4][16 * ST]; // per-wave P re-layout buffer

    // --- load Q tile 64x64 ---
#pragma unroll
    for (int it = 0; it < 8; ++it) {
        int idx = it * 128 + tid;           // 0..1023
        int row = idx >> 4;
        int dq  = (idx & 15) * 4;
        const float4 q4 = *(const float4*)(qkv + (size_t)(b * S + qt * 64 + row) * QKV + h * 64 + dq);
        u16x4 pk = {f2bf(q4.x), f2bf(q4.y), f2bf(q4.z), f2bf(q4.w)};
        *(u16x4*)(&sQ[row * ST + dq]) = pk;
    }
    __syncthreads();

    // Q A-fragments are invariant across key tiles
    v16bf qfrag[2];
#pragma unroll
    for (int kc = 0; kc < 2; ++kc) {
        const unsigned short* p = &sQ[(wave * 16 + r16) * ST + kc * 32 + 8 * half];
        qfrag[kc] = frag16(p, p + 16);
    }

    v8f o[4];
#pragma unroll
    for (int nt = 0; nt < 4; ++nt)
        o[nt] = (v8f){0.f, 0.f, 0.f, 0.f, 0.f, 0.f, 0.f, 0.f};
    float mrow[8], lrow[8];
#pragma unroll
    for (int v = 0; v < 8; ++v) { mrow[v] = -3.0e38f; lrow[v] = 0.f; }

    for (int j = 0; j <= qt; ++j) {          // causal: only key tiles <= query tile
        // --- load K tile [key][d] and V tile transposed [dh][key] ---
#pragma unroll
        for (int it = 0; it < 8; ++it) {
            int idx = it * 128 + tid;
            int row = idx >> 4;
            int dq  = (idx & 15) * 4;
            const float4 k4 = *(const float4*)(qkv + (size_t)(b * S + j * 64 + row) * QKV + D + h * 64 + dq);
            u16x4 pk = {f2bf(k4.x), f2bf(k4.y), f2bf(k4.z), f2bf(k4.w)};
            *(u16x4*)(&sK[row * ST + dq]) = pk;
            const float4 vv = *(const float4*)(qkv + (size_t)(b * S + j * 64 + row) * QKV + 2 * D + h * 64 + dq);
            sVT[(dq + 0) * ST + row] = f2bf(vv.x);
            sVT[(dq + 1) * ST + row] = f2bf(vv.y);
            sVT[(dq + 2) * ST + row] = f2bf(vv.z);
            sVT[(dq + 3) * ST + row] = f2bf(vv.w);
        }
        // prefetch next key tile's K and V lines into GL2
        if (j < qt) {
#pragma unroll
            for (int it = 0; it < 8; ++it) {
                int idx = it * 128 + tid;
                int row = idx >> 4;
                int dq  = (idx & 15) * 4;
                const float* kp = qkv + (size_t)(b * S + (j + 1) * 64 + row) * QKV + D + h * 64 + dq;
                __builtin_prefetch(kp, 0, 1);       // K
                __builtin_prefetch(kp + D, 0, 1);   // V
            }
        }
        __syncthreads();

        // --- scores S = Q K^T * 1/8, causal mask on diagonal tile ---
        v8f sacc[4];
#pragma unroll
        for (int nt = 0; nt < 4; ++nt)
            sacc[nt] = (v8f){0.f, 0.f, 0.f, 0.f, 0.f, 0.f, 0.f, 0.f};
#pragma unroll
        for (int nt = 0; nt < 4; ++nt)
#pragma unroll
            for (int kc = 0; kc < 2; ++kc) {
                const unsigned short* p = &sK[(nt * 16 + r16) * ST + kc * 32 + 8 * half];
                v16bf bf = frag16(p, p + 16);
                sacc[nt] = __builtin_amdgcn_wmma_f32_16x16x32_bf16(
                    false, qfrag[kc], false, bf, (short)0, sacc[nt], false, false);
            }
        const bool diag = (j == qt);
#pragma unroll
        for (int nt = 0; nt < 4; ++nt)
#pragma unroll
            for (int v = 0; v < 8; ++v) {
                float s = sacc[nt][v] * 0.125f;
                if (diag) {
                    int qr = wave * 16 + v + 8 * half;   // local query row 0..63
                    int kr = nt * 16 + r16;              // local key row   0..63
                    if (kr > qr) s = -1e9f;
                }
                sacc[nt][v] = s;
            }

        // --- online softmax (rows live in VGPR idx + lane-half) ---
        float rm[8];
#pragma unroll
        for (int v = 0; v < 8; ++v) {
            float m = sacc[0][v];
            m = fmaxf(m, sacc[1][v]);
            m = fmaxf(m, sacc[2][v]);
            m = fmaxf(m, sacc[3][v]);
#pragma unroll
            for (int msk = 1; msk < 16; msk <<= 1)
                m = fmaxf(m, __shfl_xor(m, msk, 32));
            rm[v] = m;
        }
        float alpha[8], newm[8];
#pragma unroll
        for (int v = 0; v < 8; ++v) {
            newm[v]  = fmaxf(mrow[v], rm[v]);
            alpha[v] = __expf(mrow[v] - newm[v]);
            mrow[v]  = newm[v];
        }
        float rs[8];
#pragma unroll
        for (int v = 0; v < 8; ++v) rs[v] = 0.f;
#pragma unroll
        for (int nt = 0; nt < 4; ++nt)
#pragma unroll
            for (int v = 0; v < 8; ++v) {
                float pv = __expf(sacc[nt][v] - newm[v]);
                sacc[nt][v] = pv;
                rs[v] += pv;
            }
#pragma unroll
        for (int v = 0; v < 8; ++v) {
            float r = rs[v];
#pragma unroll
            for (int msk = 1; msk < 16; msk <<= 1)
                r += __shfl_xor(r, msk, 32);
            lrow[v] = lrow[v] * alpha[v] + r;
        }
#pragma unroll
        for (int nt = 0; nt < 4; ++nt)
#pragma unroll
            for (int v = 0; v < 8; ++v)
                o[nt][v] *= alpha[v];

        // --- P: C-layout -> A-layout via per-wave LDS bounce (bf16) ---
#pragma unroll
        for (int nt = 0; nt < 4; ++nt)
#pragma unroll
            for (int v = 0; v < 8; ++v)
                sP[wave][(v + 8 * half) * ST + nt * 16 + r16] = f2bf(sacc[nt][v]);

        // --- O += P @ V ---
#pragma unroll
        for (int kc = 0; kc < 2; ++kc) {
            const unsigned short* pp = &sP[wave][r16 * ST + kc * 32 + 8 * half];
            v16bf pfrag = frag16(pp, pp + 16);
#pragma unroll
            for (int nt = 0; nt < 4; ++nt) {
                const unsigned short* vp = &sVT[(nt * 16 + r16) * ST + kc * 32 + 8 * half];
                v16bf bf = frag16(vp, vp + 16);
                o[nt] = __builtin_amdgcn_wmma_f32_16x16x32_bf16(
                    false, pfrag, false, bf, (short)0, o[nt], false, false);
            }
        }
        __syncthreads();   // protect sK/sVT for next key tile
    }

    // --- normalize + store [token][h*64+dh] ---
#pragma unroll
    for (int nt = 0; nt < 4; ++nt)
#pragma unroll
        for (int v = 0; v < 8; ++v) {
            int token = b * S + qt * 64 + wave * 16 + v + 8 * half;
            int col   = h * 64 + nt * 16 + r16;
            out[(size_t)token * D + col] = o[nt][v] / lrow[v];
        }
}

// ---------------------------------------------------------------------------
// Residual + LayerNorm (matches reference: unbiased var (N-1), eps on std)
// One block per token row, 256 threads (8 waves), D=1024.
// wave32 shuffle reductions + one 8-element cross-wave combine.
// ---------------------------------------------------------------------------
__global__ __launch_bounds__(256)
void ln_kernel(const float* __restrict__ xa, const float* __restrict__ xb,
               const float* __restrict__ g, const float* __restrict__ bt,
               float* __restrict__ out) {
    constexpr int D = 1024;
    const int row  = blockIdx.x;
    const int tid  = threadIdx.x;
    const int wave = tid >> 5;
    const int lane = tid & 31;
    __shared__ float sred[8];

    const float4 a4 = *(const float4*)(xa + (size_t)row * D + tid * 4);
    const float4 b4 = *(const float4*)(xb + (size_t)row * D + tid * 4);
    float v[4] = {a4.x + b4.x, a4.y + b4.y, a4.z + b4.z, a4.w + b4.w};

    float s = v[0] + v[1] + v[2] + v[3];
#pragma unroll
    for (int msk = 1; msk < 32; msk <<= 1) s += __shfl_xor(s, msk, 32);
    if (lane == 0) sred[wave] = s;
    __syncthreads();
    float tot = sred[0] + sred[1] + sred[2] + sred[3] +
                sred[4] + sred[5] + sred[6] + sred[7];
    float mean = tot * (1.0f / D);

    float sq = 0.f;
#pragma unroll
    for (int i = 0; i < 4; ++i) { float d = v[i] - mean; sq += d * d; }
    __syncthreads();                       // sred reuse
#pragma unroll
    for (int msk = 1; msk < 32; msk <<= 1) sq += __shfl_xor(sq, msk, 32);
    if (lane == 0) sred[wave] = sq;
    __syncthreads();
    float tot2 = sred[0] + sred[1] + sred[2] + sred[3] +
                 sred[4] + sred[5] + sred[6] + sred[7];
    float var = tot2 * (1.0f / (D - 1));
    float inv = 1.0f / (sqrtf(var) + 1e-6f);

    const float4 g4  = *(const float4*)(g  + tid * 4);
    const float4 bb4 = *(const float4*)(bt + tid * 4);
    float4 o4;
    o4.x = g4.x * (v[0] - mean) * inv + bb4.x;
    o4.y = g4.y * (v[1] - mean) * inv + bb4.y;
    o4.z = g4.z * (v[2] - mean) * inv + bb4.z;
    o4.w = g4.w * (v[3] - mean) * inv + bb4.w;
    *(float4*)(out + (size_t)row * D + tid * 4) = o4;
}

// ---------------------------------------------------------------------------
// Launch: QKV GEMM -> flash attn -> aproj GEMM -> LN1 -> FC+GELU -> mproj -> LN2
// ---------------------------------------------------------------------------
extern "C" void kernel_launch(void* const* d_in, const int* in_sizes, int n_in,
                              void* d_out, int out_size, void* d_ws, size_t ws_size,
                              hipStream_t stream) {
    const float* x       = (const float*)d_in[0];
    const float* w_attn  = (const float*)d_in[1];
    const float* b_attn  = (const float*)d_in[2];
    const float* w_aproj = (const float*)d_in[3];
    const float* b_aproj = (const float*)d_in[4];
    const float* g1      = (const float*)d_in[5];
    const float* b1      = (const float*)d_in[6];
    const float* w_fc    = (const float*)d_in[7];
    const float* b_fc    = (const float*)d_in[8];
    const float* w_mproj = (const float*)d_in[9];
    const float* b_mproj = (const float*)d_in[10];
    const float* g2      = (const float*)d_in[11];
    const float* b2      = (const float*)d_in[12];
    float* out = (float*)d_out;
    float* ws  = (float*)d_ws;

    const int Ntok = 4096;                 // B*S
    // workspace layout (floats), with h overlaying the dead qkv+a0 region
    float* qkv  = ws;                      // 4096*3072 = 12582912
    float* a0   = ws + 12582912u;          // 4096*1024 =  4194304
    float* hbuf = ws;                      // 4096*4096 = 16777216 (overlays qkv+a0)
    float* aout = ws + 16777216u;          // 4194304
    float* n1   = ws + 20971520u;          // 4194304
    float* mb   = ws + 25165824u;          // 4194304

    dim3 blk(256);
    gemm_bias<false><<<dim3(3072 / 128, Ntok / 128), blk, 0, stream>>>(
        x, w_attn, b_attn, qkv, 1024, 3072);
    attn_kernel<<<dim3(32, 32), dim3(128), 0, stream>>>(qkv, a0);
    gemm_bias<false><<<dim3(1024 / 128, Ntok / 128), blk, 0, stream>>>(
        a0, w_aproj, b_aproj, aout, 1024, 1024);
    ln_kernel<<<dim3(Ntok), blk, 0, stream>>>(x, aout, g1, b1, n1);
    gemm_bias<true><<<dim3(4096 / 128, Ntok / 128), blk, 0, stream>>>(
        n1, w_fc, b_fc, hbuf, 1024, 4096);
    gemm_bias<false><<<dim3(1024 / 128, Ntok / 128), blk, 0, stream>>>(
        hbuf, w_mproj, b_mproj, mb, 4096, 1024);
    ln_kernel<<<dim3(Ntok), blk, 0, stream>>>(n1, mb, g2, b2, out);
}
